// ChessNN_64518998721049
// MI455X (gfx1250) — matrix operand order repaired and checked by reading, not compiled, not run on hardware
//
#include <hip/hip_runtime.h>
#include <hip/hip_bf16.h>
#include <stdint.h>

// ---------------------------------------------------------------------------
// NNUE forward for MI455X (gfx1250, wave32, WMMA).
//   L0 (dominant): X[4096,41024] @ W0^T[41024,256] for white & black.
//   Roofline: 1.34 GB feature stream @ 23.3 TB/s ~ 58 us (HBM floor).
//   f16 WMMA (v_wmma_f32_16x16x32_f16, f32 accumulate) keeps compute under it.
//   WG shape 64 rows x 128 cols halves L2->WGP W0 traffic.
//   Inner loop: 2-deep ping-pong pipeline; sched_barrier(0) pins each
//   half-step's loads ahead of the other half's WMMAs so the allocator keeps
//   BOTH full B-fragment buffers live (no load->s_wait 0->wmma serialization).
//   Single B base pointer + immediate tile offsets (j*16*F*2 < 2^23).
// ---------------------------------------------------------------------------

typedef __attribute__((ext_vector_type(16))) _Float16 v16h;
typedef __attribute__((ext_vector_type(4)))  _Float16 v4h;
typedef __attribute__((ext_vector_type(8)))  float    v8f;

#define NNUE_F 41024   // input features
#define NNUE_M 256     // L0 width
#define NNUE_B 4096    // batch

// ---------------------------------------------------------------------------
// Kernel 0: convert W0 (fp32, [256,41024]) -> f16 in workspace (21 MB, stays
// in the 192 MB L2). 4 elements per thread, vectorized.
// ---------------------------------------------------------------------------
__global__ void __launch_bounds__(256)
nnue_cvt_w0(const float* __restrict__ W0, _Float16* __restrict__ W0h, int n4) {
    int i = blockIdx.x * blockDim.x + threadIdx.x;
    if (i < n4) {
        float4 f = ((const float4*)W0)[i];
        v4h h;
        h.x = (_Float16)f.x; h.y = (_Float16)f.y;
        h.z = (_Float16)f.z; h.w = (_Float16)f.w;
        ((v4h*)W0h)[i] = h;
    }
}

// ---------------------------------------------------------------------------
// Fragment buffer for one 32-wide K-step: raw fp32 A data + 4 f16 B tiles.
// A-fragment (16-bit A 16x32, contiguous k, as B): lane L (L<16) = row L,
// K = kb..kb+15; lane L+16 = row L, K = kb+16..kb+31.
// B-fragment (32x16): lane = column, lanes 0-15 hold
// K=kb..kb+15 contiguous, lanes 16-31 hold K=kb+16..kb+31.
// ---------------------------------------------------------------------------
struct NnueFrag {
    float4 f0, f1, f2, f3;
    v16h   b0, b1, b2, b3;
};

__device__ __forceinline__ void nnue_frag_load(
    NnueFrag& g, const float* __restrict__ xrow,
    const _Float16* __restrict__ wbase, int kb)
{
    const float* xp = xrow + kb;
    g.f0 = *(const float4*)(xp);
    g.f1 = *(const float4*)(xp + 4);
    g.f2 = *(const float4*)(xp + 8);
    g.f3 = *(const float4*)(xp + 12);
    // single base + immediate tile offsets (16*NNUE_F elements apart)
    g.b0 = *(const v16h*)(wbase + kb);
    g.b1 = *(const v16h*)(wbase + kb + 16 * NNUE_F);
    g.b2 = *(const v16h*)(wbase + kb + 32 * NNUE_F);
    g.b3 = *(const v16h*)(wbase + kb + 48 * NNUE_F);
}

__device__ __forceinline__ void nnue_frag_mma(
    const NnueFrag& g, v8f& acc0, v8f& acc1, v8f& acc2, v8f& acc3)
{
    v16h a;
    a[0]  = (_Float16)g.f0.x; a[1]  = (_Float16)g.f0.y;
    a[2]  = (_Float16)g.f0.z; a[3]  = (_Float16)g.f0.w;
    a[4]  = (_Float16)g.f1.x; a[5]  = (_Float16)g.f1.y;
    a[6]  = (_Float16)g.f1.z; a[7]  = (_Float16)g.f1.w;
    a[8]  = (_Float16)g.f2.x; a[9]  = (_Float16)g.f2.y;
    a[10] = (_Float16)g.f2.z; a[11] = (_Float16)g.f2.w;
    a[12] = (_Float16)g.f3.x; a[13] = (_Float16)g.f3.y;
    a[14] = (_Float16)g.f3.z; a[15] = (_Float16)g.f3.w;

    acc0 = __builtin_amdgcn_wmma_f32_16x16x32_f16(
               false, a, false, g.b0, (short)0, acc0, false, false);
    acc1 = __builtin_amdgcn_wmma_f32_16x16x32_f16(
               false, a, false, g.b1, (short)0, acc1, false, false);
    acc2 = __builtin_amdgcn_wmma_f32_16x16x32_f16(
               false, a, false, g.b2, (short)0, acc2, false, false);
    acc3 = __builtin_amdgcn_wmma_f32_16x16x32_f16(
               false, a, false, g.b3, (short)0, acc3, false, false);
}

// ---------------------------------------------------------------------------
// Kernel 1: L0 GEMM via v_wmma_f32_16x16x32_f16.
//   grid = (B/64 row blocks, 2 col halves, 2 perspectives), 256 threads.
//   8 waves = 4 batch quarters (16 rows) x 2 col tiles (64 cols).
//   Each wave: 16 rows x 64 cols = 4 accumulator tiles; ping-pong pipeline.
// ---------------------------------------------------------------------------
__global__ void __launch_bounds__(256)
__attribute__((amdgpu_waves_per_eu(1)))
nnue_l0_gemm(const float* __restrict__ Xw, const float* __restrict__ Xb,
             const _Float16* __restrict__ W0h,
             float* __restrict__ Ow, float* __restrict__ Ob)
{
    const float* __restrict__ X = blockIdx.z ? Xb : Xw;
    float* __restrict__       O = blockIdx.z ? Ob : Ow;

    const int lane = threadIdx.x & 31;
    const int wave = threadIdx.x >> 5;
    const int bq   = wave & 3;        // batch quarter (16 rows of 64)
    const int nh   = wave >> 2;       // col tile (64 cols of 128)
    const int hi   = lane >> 4;       // lane half
    const int l15  = lane & 15;

    const int rowbase = blockIdx.x * 64 + bq * 16;
    const int nbase   = blockIdx.y * 128 + nh * 64;
    const int r       = rowbase + l15;

    // Per-lane base pointers (hoisted out of the K loop).
    const float* xrow = X + (size_t)r * NNUE_F + hi * 16;
    const _Float16* wbase = W0h + (size_t)(nbase + l15) * NNUE_F + hi * 16;

    v8f acc0 = {}, acc1 = {}, acc2 = {}, acc3 = {};

    // ---- ping-pong pipeline: NNUE_F = 64*641, so the 64-wide loop is exact.
    NnueFrag pa, pb;
    nnue_frag_load(pa, xrow, wbase, 0);

    int kb = 0;
    for (; kb < NNUE_F - 64; kb += 64) {
        nnue_frag_load(pb, xrow, wbase, kb + 32);
        __builtin_amdgcn_sched_barrier(0);   // pin pb loads before pa's WMMAs
        nnue_frag_mma(pa, acc0, acc1, acc2, acc3);          // step kb
        nnue_frag_load(pa, xrow, wbase, kb + 64);
        __builtin_amdgcn_sched_barrier(0);   // pin pa loads before pb's WMMAs
        nnue_frag_mma(pb, acc0, acc1, acc2, acc3);          // step kb+32
    }
    // tail: kb == NNUE_F-64 here; pa holds step kb.
    nnue_frag_load(pb, xrow, wbase, kb + 32);
    nnue_frag_mma(pa, acc0, acc1, acc2, acc3);              // step F-64
    nnue_frag_mma(pb, acc0, acc1, acc2, acc3);              // step F-32

    // ---- store: C/D layout: VGPR i -> row (hi*8 + i), col = lane&15 -------
    const int orow  = rowbase + hi * 8;
    const int ocol0 = nbase + l15;
#pragma unroll
    for (int i = 0; i < 8; ++i) {
        float* op = O + (size_t)(orow + i) * NNUE_M + ocol0;
        op[0]  = acc0[i];
        op[16] = acc1[i];
        op[32] = acc2[i];
        op[48] = acc3[i];
    }
}

// ---------------------------------------------------------------------------
// Kernel 2: stm blend + clip + small layers (W1:[32,512], W2:[32,32],
// W3:[1,32]). 8 rows per block; 32 lanes per row (lane = neuron).
// ---------------------------------------------------------------------------
__device__ __forceinline__ float nnue_clip01(float x) {
    return fminf(fmaxf(x, 0.0f), 1.0f);
}

__global__ void __launch_bounds__(256)
nnue_head(const float* __restrict__ wacc, const float* __restrict__ bacc,
          const float* __restrict__ stm,  const float* __restrict__ b0,
          const float* __restrict__ W1,   const float* __restrict__ b1,
          const float* __restrict__ W2,   const float* __restrict__ b2,
          const float* __restrict__ W3,   const float* __restrict__ b3,
          float* __restrict__ out)
{
    __shared__ float l1[8][512];
    __shared__ float l2[8][32];
    __shared__ float l3[8][32];

    const int lane = threadIdx.x & 31;
    const int rl   = threadIdx.x >> 5;
    const int row  = blockIdx.x * 8 + rl;
    const float s  = stm[row];

    // acc = stm*[w,b] + (1-stm)*[b,w], then clip -> l1
    for (int i = lane; i < NNUE_M; i += 32) {
        float w = wacc[(size_t)row * NNUE_M + i] + b0[i];
        float b = bacc[(size_t)row * NNUE_M + i] + b0[i];
        l1[rl][i]          = nnue_clip01(s * w + (1.0f - s) * b);
        l1[rl][NNUE_M + i] = nnue_clip01(s * b + (1.0f - s) * w);
    }
    __syncthreads();

    // layer 1: [512] -> [32]
    float a1 = b1[lane];
    const float* w1r = W1 + lane * 512;
#pragma unroll 4
    for (int k = 0; k < 512; ++k) a1 += l1[rl][k] * w1r[k];
    l2[rl][lane] = nnue_clip01(a1);
    __syncthreads();

    // layer 2: [32] -> [32]
    float a2 = b2[lane];
    const float* w2r = W2 + lane * 32;
#pragma unroll
    for (int k = 0; k < 32; ++k) a2 += l2[rl][k] * w2r[k];
    l3[rl][lane] = nnue_clip01(a2);
    __syncthreads();

    // layer 3: [32] -> [1], wave32 shuffle reduction
    float t = l3[rl][lane] * W3[lane];
    for (int off = 16; off; off >>= 1) t += __shfl_down(t, off);
    if (lane == 0) out[row] = t + b3[0];
}

// ---------------------------------------------------------------------------
// Launch: inputs in setup_inputs() order.
// Workspace: [W0h f16: 21,004,288 B][wacc f32: 4,194,304 B][bacc f32: 4,194,304 B]
// ---------------------------------------------------------------------------
extern "C" void kernel_launch(void* const* d_in, const int* in_sizes, int n_in,
                              void* d_out, int out_size, void* d_ws, size_t ws_size,
                              hipStream_t stream)
{
    const float* Xw  = (const float*)d_in[0];
    const float* Xb  = (const float*)d_in[1];
    const float* stm = (const float*)d_in[2];
    const float* W0  = (const float*)d_in[3];
    const float* b0  = (const float*)d_in[4];
    const float* W1  = (const float*)d_in[5];
    const float* b1  = (const float*)d_in[6];
    const float* W2  = (const float*)d_in[7];
    const float* b2  = (const float*)d_in[8];
    const float* W3  = (const float*)d_in[9];
    const float* b3  = (const float*)d_in[10];
    float* out = (float*)d_out;

    char* ws = (char*)d_ws;
    _Float16* W0h = (_Float16*)ws;                                    // 21,004,288 B
    float*    wacc = (float*)(ws + (size_t)21004288);                 //  4,194,304 B
    float*    bacc = (float*)(ws + (size_t)21004288 + 4194304);       //  4,194,304 B

    const int n4 = (NNUE_M * NNUE_F) / 4;
    nnue_cvt_w0<<<(n4 + 255) / 256, 256, 0, stream>>>(W0, W0h, n4);

    nnue_l0_gemm<<<dim3(NNUE_B / 64, 2, 2), 256, 0, stream>>>(Xw, Xb, W0h, wacc, bacc);

    nnue_head<<<NNUE_B / 8, 256, 0, stream>>>(wacc, bacc, stm, b0, W1, b1,
                                              W2, b2, W3, b3, out);
}
